// GeoChebyConv_893353198326
// MI455X (gfx1250) — compile-verified, matrix operand order checked by reading, not run on hardware
//
#include <hip/hip_runtime.h>
#include <hip/hip_bf16.h>
#include <math.h>

// ---- problem constants (match reference) ----
#define GN 100000      // nodes  (== 6250 * 16, exact 16-row tiles)
#define GE 1600000     // edges
#define GF 64          // input features
#define GH 64          // hidden
#define GC 16          // classes

typedef __attribute__((ext_vector_type(16))) _Float16 v16h;
typedef __attribute__((ext_vector_type(8)))  float    v8f;

// ---------------------------------------------------------------------------
// elementwise / graph kernels
// ---------------------------------------------------------------------------
__global__ void zero_kernel(float* p, int n) {
    int i = blockIdx.x * blockDim.x + threadIdx.x;
    int stride = gridDim.x * blockDim.x;
    for (; i < n; i += stride) p[i] = 0.0f;
}

__global__ void degree_kernel(const int* __restrict__ src,
                              const float* __restrict__ w,
                              float* __restrict__ deg) {
    int e = blockIdx.x * blockDim.x + threadIdx.x;
    if (e < GE) unsafeAtomicAdd(&deg[src[e]], w[e]);
}

__global__ void dinv_kernel(float* deg) {
    int i = blockIdx.x * blockDim.x + threadIdx.x;
    if (i < GN) {
        float d = deg[i];
        deg[i] = (d > 0.0f) ? rsqrtf(fmaxf(d, 1e-12f)) : 0.0f;
    }
}

__global__ void normw_kernel(const int* __restrict__ src,
                             const int* __restrict__ dst,
                             const float* __restrict__ w,
                             const float* __restrict__ dinv,
                             float* __restrict__ normw) {
    int e = blockIdx.x * blockDim.x + threadIdx.x;
    if (e < GE) normw[e] = -dinv[src[e]] * w[e] * dinv[dst[e]];
}

// one thread handles one edge x 4 features (float4 load, 4 L2 atomics)
__global__ void scatter_kernel(const int* __restrict__ src,
                               const int* __restrict__ dst,
                               const float* __restrict__ normw,
                               const float* __restrict__ x,
                               float* __restrict__ tx) {
    int t = blockIdx.x * blockDim.x + threadIdx.x;   // < GE * 16
    int e = t >> 4;
    int c = (t & 15) << 2;
    if (e < GE) {
        float wv = normw[e];
        const float4 v = *(const float4*)(x + (size_t)src[e] * GF + c);
        float* td = tx + (size_t)dst[e] * GF + c;
        unsafeAtomicAdd(td + 0, wv * v.x);
        unsafeAtomicAdd(td + 1, wv * v.y);
        unsafeAtomicAdd(td + 2, wv * v.z);
        unsafeAtomicAdd(td + 3, wv * v.w);
    }
}

// ---------------------------------------------------------------------------
// Weight pre-swizzle: f32 (KxN row-major) -> f16 WMMA B-fragments laid out so
// a lane's 16 halves are 32 contiguous bytes.  ISA 7.12.2 B layout (32x16):
// lane = N column (0-15 | 16-31 repeat), kb = (lane&16)?8:0, halves j<8 hold
// K=k0+kb+j, halves j>=8 hold K=k0+kb+16+(j-8).
// Fragment index: (kt*ntiles + nt)*32 + lane.
// ---------------------------------------------------------------------------
__global__ void wprep_kernel(const float* __restrict__ W, int ld,
                             int ktiles, int ntiles, _Float16* __restrict__ out) {
    int t = blockIdx.x * blockDim.x + threadIdx.x;
    if (t >= ktiles * ntiles * 32) return;
    const int lane = t & 31;
    const int frag = t >> 5;
    const int nt = frag % ntiles;
    const int kt = frag / ntiles;
    const int nc = nt * 16 + (lane & 15);
    const int k0 = kt * 32 + ((lane & 16) ? 8 : 0);
    _Float16* o = out + (size_t)t * 16;
#pragma unroll
    for (int j = 0; j < 8; ++j) {
        o[j]     = (_Float16)W[(size_t)(k0 + j) * ld + nc];
        o[j + 8] = (_Float16)W[(size_t)(k0 + 16 + j) * ld + nc];
    }
}

__device__ inline v16h load_wfrag(const _Float16* __restrict__ base, int ntiles,
                                  int kt, int nt, int lane) {
    return *(const v16h*)(base + (size_t)(((kt * ntiles + nt) << 5) + lane) * 16);
}

// A fragment from f32 matrix (16x32 tile, convert in-register)
__device__ inline v16h load_a_frag(const float* __restrict__ base, int ld,
                                   int row0, int k0, int lane) {
    const int lr = lane & 15;
    const int kb = (lane & 16) ? 8 : 0;
    const float* p = base + (size_t)(row0 + lr) * ld + k0 + kb;
    v16h a;
#pragma unroll
    for (int j = 0; j < 8; ++j) {
        a[j]     = (_Float16)p[j];
        a[j + 8] = (_Float16)p[16 + j];
    }
    return a;
}

__device__ inline v8f wmma_f16(v16h a, v16h b, v8f c) {
    return __builtin_amdgcn_wmma_f32_16x16x32_f16(
        /*neg_a=*/false, a, /*neg_b=*/false, b,
        /*c_mod=*/(short)0, c, /*reuse_a=*/false, /*reuse_b=*/false);
}

// ---------------------------------------------------------------------------
// Layer 1: h = relu(x @ W0 + tx @ W1 + b)   (N x 64)
// One wave computes a full 16x64 row strip: 4 accumulators, A fragments
// loaded once and reused across the 4 column tiles -> 16 WMMAs per wave.
// ---------------------------------------------------------------------------
__global__ __launch_bounds__(256) void gemm1_kernel(
        const float* __restrict__ x, const float* __restrict__ tx,
        const _Float16* __restrict__ wf0, const _Float16* __restrict__ wf1,
        const float* __restrict__ bias, float* __restrict__ h) {
    const int wave = (blockIdx.x * blockDim.x + threadIdx.x) >> 5;
    const int lane = threadIdx.x & 31;
    if (wave >= GN / 16) return;            // wave-uniform
    const int row0 = wave * 16;

    v8f acc[4] = {{}, {}, {}, {}};
#pragma unroll
    for (int kt = 0; kt < 2; ++kt) {        // K = 0, 32
        const v16h ax = load_a_frag(x,  GF, row0, kt * 32, lane);
        const v16h at = load_a_frag(tx, GF, row0, kt * 32, lane);
#pragma unroll
        for (int nt = 0; nt < 4; ++nt) {
            acc[nt] = wmma_f16(ax, load_wfrag(wf0, 4, kt, nt, lane), acc[nt]);
            acc[nt] = wmma_f16(at, load_wfrag(wf1, 4, kt, nt, lane), acc[nt]);
        }
    }

    const int m0 = row0 + ((lane & 16) ? 8 : 0);
#pragma unroll
    for (int nt = 0; nt < 4; ++nt) {
        const int n = nt * 16 + (lane & 15);
        const float bv = bias[n];
#pragma unroll
        for (int v = 0; v < 8; ++v)
            h[(size_t)(m0 + v) * GH + n] = fmaxf(acc[nt][v] + bv, 0.0f);
    }
}

// ---------------------------------------------------------------------------
// Layer 2: out = h @ W2_0 + tx2 @ W2_1 + b2   (N x 16); one tile per wave
// ---------------------------------------------------------------------------
__global__ __launch_bounds__(256) void gemm2_kernel(
        const float* __restrict__ h, const float* __restrict__ tx,
        const _Float16* __restrict__ wf0, const _Float16* __restrict__ wf1,
        const float* __restrict__ bias, float* __restrict__ out) {
    const int wave = (blockIdx.x * blockDim.x + threadIdx.x) >> 5;
    const int lane = threadIdx.x & 31;
    if (wave >= GN / 16) return;            // wave-uniform
    const int row0 = wave * 16;

    v8f acc = {};
#pragma unroll
    for (int kt = 0; kt < 2; ++kt) {
        acc = wmma_f16(load_a_frag(h,  GH, row0, kt * 32, lane),
                       load_wfrag(wf0, 1, kt, 0, lane), acc);
        acc = wmma_f16(load_a_frag(tx, GH, row0, kt * 32, lane),
                       load_wfrag(wf1, 1, kt, 0, lane), acc);
    }

    const int n = lane & 15;
    const int m0 = row0 + ((lane & 16) ? 8 : 0);
    const float bv = bias[n];
#pragma unroll
    for (int v = 0; v < 8; ++v)
        out[(size_t)(m0 + v) * GC + n] = acc[v] + bv;
}

// ---------------------------------------------------------------------------
// row-wise log_softmax over 16 classes, in place
// ---------------------------------------------------------------------------
__global__ void log_softmax_kernel(float* out) {
    int i = blockIdx.x * blockDim.x + threadIdx.x;
    if (i >= GN) return;
    float* row = out + (size_t)i * GC;
    float r[GC];
    float m = -INFINITY;
#pragma unroll
    for (int j = 0; j < GC; ++j) { r[j] = row[j]; m = fmaxf(m, r[j]); }
    float s = 0.0f;
#pragma unroll
    for (int j = 0; j < GC; ++j) s += __expf(r[j] - m);
    const float lse = m + __logf(s);
#pragma unroll
    for (int j = 0; j < GC; ++j) row[j] = r[j] - lse;
}

// ---------------------------------------------------------------------------
extern "C" void kernel_launch(void* const* d_in, const int* in_sizes, int n_in,
                              void* d_out, int out_size, void* d_ws, size_t ws_size,
                              hipStream_t stream) {
    const float* feat = (const float*)d_in[0];
    const int*   eidx = (const int*)d_in[1];
    const float* ew   = (const float*)d_in[2];
    const float* W1_0 = (const float*)d_in[3];
    const float* W1_1 = (const float*)d_in[4];
    const float* b1   = (const float*)d_in[5];
    const float* W2_0 = (const float*)d_in[6];
    const float* W2_1 = (const float*)d_in[7];
    const float* b2   = (const float*)d_in[8];
    const int* src = eidx;        // edge_index[0]
    const int* dst = eidx + GE;   // edge_index[1]
    float* out = (float*)d_out;

    // workspace layout (floats): deg[N] | normw[E] | tx[N*64] | h[N*64] | wf16
    float* deg   = (float*)d_ws;
    float* normw = deg + GN;
    float* tx    = normw + GE;
    float* h     = tx + (size_t)GN * GF;
    _Float16* wf1_0 = (_Float16*)(h + (size_t)GN * GH);   // 2kt*4nt*32*16 = 4096
    _Float16* wf1_1 = wf1_0 + 4096;
    _Float16* wf2_0 = wf1_1 + 4096;                       // 2kt*1nt*32*16 = 1024
    _Float16* wf2_1 = wf2_0 + 1024;

    const int T = 256;
    // 0) pre-swizzle weights into f16 WMMA fragments (tiny, once per call)
    wprep_kernel<<<1, 256, 0, stream>>>(W1_0, GH, 2, 4, wf1_0);
    wprep_kernel<<<1, 256, 0, stream>>>(W1_1, GH, 2, 4, wf1_1);
    wprep_kernel<<<1, 64,  0, stream>>>(W2_0, GC, 2, 1, wf2_0);
    wprep_kernel<<<1, 64,  0, stream>>>(W2_1, GC, 2, 1, wf2_1);
    // 1) zero accumulators
    zero_kernel<<<1024, T, 0, stream>>>(deg, GN);
    zero_kernel<<<4096, T, 0, stream>>>(tx, GN * GF);
    // 2) degree + dinv + norm_w
    degree_kernel<<<(GE + T - 1) / T, T, 0, stream>>>(src, ew, deg);
    dinv_kernel<<<(GN + T - 1) / T, T, 0, stream>>>(deg);
    normw_kernel<<<(GE + T - 1) / T, T, 0, stream>>>(src, dst, ew, deg, normw);
    // 3) tx1 = scatter(norm_w * x[src] -> dst)
    scatter_kernel<<<(GE * 16) / T, T, 0, stream>>>(src, dst, normw, feat, tx);
    // 4) h = relu(x@W1_0 + tx1@W1_1 + b1)   [WMMA, 16 per wave]
    gemm1_kernel<<<(GN / 16 + 7) / 8, T, 0, stream>>>(feat, tx, wf1_0, wf1_1, b1, h);
    // 5) tx2 = scatter(norm_w * h[src] -> dst)
    zero_kernel<<<4096, T, 0, stream>>>(tx, GN * GH);
    scatter_kernel<<<(GE * 16) / T, T, 0, stream>>>(src, dst, normw, h, tx);
    // 6) logits = h@W2_0 + tx2@W2_1 + b2    [WMMA]
    gemm2_kernel<<<(GN / 16 + 7) / 8, T, 0, stream>>>(h, tx, wf2_0, wf2_1, b2, out);
    // 7) log_softmax rows
    log_softmax_kernel<<<(GN + T - 1) / T, T, 0, stream>>>(out);
}
